// PointGroup_2508260901476
// MI455X (gfx1250) — compile-verified, matrix operand order checked by reading, not run on hardware
//
#include <hip/hip_runtime.h>
#include <hip/hip_bf16.h>
#include <math.h>

// ---------------------------------------------------------------------------
// PointGroup loss on MI455X (gfx1250, wave32, WMMA).
// Bandwidth-bound: feat (256MB fp32) is streamed twice (stats pass + loss
// pass); all per-row math is fused per 16-row tile via v_wmma_f32_16x16x32_f16.
// L2 management: first 160MB of feat loaded RT (stays resident in the 192MB
// global L2 across both passes); the 96MB tail + all pass-2-only streams use
// non-temporal hints so they never displace the resident region.
// b1 is omitted on purpose: BatchNorm subtracts the batch mean, so a constant
// per-channel bias cancels exactly in (h - mean).
// ---------------------------------------------------------------------------

typedef __attribute__((ext_vector_type(16))) _Float16 v16h;
typedef __attribute__((ext_vector_type(8)))  _Float16 v8h;
typedef __attribute__((ext_vector_type(8)))  float    v8f;
typedef __attribute__((ext_vector_type(4)))  float    vf4;

#define N_PTS    1048576
#define N_TILES  (N_PTS / 16)
#define T_CACHE  40960        // tiles kept L2-resident: 40960*16*64*4B = 160MB
#define CC       64
#define KCLS     20
#define H_STRIDE 80   // halves; 160B rows keep ds_load_b128 16B-aligned
#define L_STRIDE 36   // floats
#define B_STRIDE 20   // floats

#define WMMA_F16(A, B, C) \
  __builtin_amdgcn_wmma_f32_16x16x32_f16(false, (A), false, (B), (short)0, (C), false, false)

template <bool NT>
__device__ __forceinline__ vf4 ld4(const float* __restrict__ p) {
  if (NT) return __builtin_nontemporal_load((const vf4*)p);
  return *(const vf4*)p;
}

// A-fragment (16x32 f16) from a global fp32 row chunk.
// Lane l: row = l&15, kb = (l>>4)*8; halves 0..7 = K[kb..kb+7],
// halves 8..15 = K[kb+16..kb+23] (ISA 7.12.2 16-bit A layout).
template <bool NT>
__device__ __forceinline__ v16h make_afrag(const float* __restrict__ p) {
  vf4 x0 = ld4<NT>(p + 0);
  vf4 x1 = ld4<NT>(p + 4);
  vf4 x2 = ld4<NT>(p + 16);
  vf4 x3 = ld4<NT>(p + 20);
  v16h a;
#pragma unroll
  for (int e = 0; e < 4; ++e) {
    a[e]      = (_Float16)x0[e];
    a[4 + e]  = (_Float16)x1[e];
    a[8 + e]  = (_Float16)x2[e];
    a[12 + e] = (_Float16)x3[e];
  }
  return a;
}

// Same fragment shape but sourced from f16 LDS (two ds_load_b128).
__device__ __forceinline__ v16h make_hfrag(const _Float16* __restrict__ p) {
  v8h lo = *(const v8h*)(p);
  v8h hi = *(const v8h*)(p + 16);
  v16h a;
#pragma unroll
  for (int e = 0; e < 8; ++e) { a[e] = lo[e]; a[8 + e] = hi[e]; }
  return a;
}

__device__ __forceinline__ float wave_sum(float v) {
#pragma unroll
  for (int off = 16; off >= 1; off >>= 1) v += __shfl_xor(v, off, 32);
  return v;
}

// ---------------------------------------------------------------------------
__global__ void init_ws(float* __restrict__ wsf) {
  const int i = threadIdx.x;
  if (i < 128) wsf[i] = 0.0f;          // col sums / sumsq of g
  if (i < 8)   wsf[256 + i] = 0.0f;    // loss accumulators
}

// Pre-swizzle W1 / Wseg / W2 (fp32 -> f16) into exact B-fragment layout so
// compute waves load them with coalesced b128 loads (L2-resident).
__global__ void prep_frags(const float* __restrict__ W1, const float* __restrict__ Wseg,
                           const float* __restrict__ W2, _Float16* __restrict__ frags) {
  const int lane = threadIdx.x & 31;
  const int kb = (lane >> 4) * 8;
  const int n0 = lane & 15;
#pragma unroll
  for (int ct = 0; ct < 4; ++ct)
#pragma unroll
    for (int ks = 0; ks < 2; ++ks) {
      _Float16* dst = frags + (ct * 2 + ks) * 512 + lane * 16;
      const int n = ct * 16 + n0;
#pragma unroll
      for (int e = 0; e < 8; ++e) {
        dst[e]     = (_Float16)W1[(ks * 32 + kb + e) * CC + n];
        dst[8 + e] = (_Float16)W1[(ks * 32 + kb + 16 + e) * CC + n];
      }
    }
#pragma unroll
  for (int ct = 0; ct < 2; ++ct)
#pragma unroll
    for (int ks = 0; ks < 2; ++ks) {
      _Float16* dst = frags + 4096 + (ct * 2 + ks) * 512 + lane * 16;
      const int n = ct * 16 + n0;
#pragma unroll
      for (int e = 0; e < 8; ++e) {
        const int k0 = ks * 32 + kb + e, k1 = k0 + 16;
        dst[e]     = (n < KCLS) ? (_Float16)Wseg[k0 * KCLS + n] : (_Float16)0.0f;
        dst[8 + e] = (n < KCLS) ? (_Float16)Wseg[k1 * KCLS + n] : (_Float16)0.0f;
      }
    }
#pragma unroll
  for (int ks = 0; ks < 2; ++ks) {
    _Float16* dst = frags + 6144 + ks * 512 + lane * 16;
#pragma unroll
    for (int e = 0; e < 8; ++e) {
      const int k0 = ks * 32 + kb + e, k1 = k0 + 16;
      dst[e]     = (n0 < 3) ? (_Float16)W2[k0 * 3 + n0] : (_Float16)0.0f;
      dst[8 + e] = (n0 < 3) ? (_Float16)W2[k1 * 3 + n0] : (_Float16)0.0f;
    }
  }
}

// Pass 1: per-channel sum / sumsq of g = feat @ W1 (WMMA tiles).
__global__ __launch_bounds__(128) void pass1_stats(const float* __restrict__ feat,
                                                   const _Float16* __restrict__ frags,
                                                   float* __restrict__ sums) {
  const int lane = threadIdx.x & 31;
  const int gw = (blockIdx.x * blockDim.x + threadIdx.x) >> 5;
  const int nw = (gridDim.x * blockDim.x) >> 5;
  const int r = lane & 15;
  const int kb = (lane >> 4) * 8;

  v16h B1[8];
#pragma unroll
  for (int f = 0; f < 8; ++f) B1[f] = *(const v16h*)(frags + f * 512 + lane * 16);

  float acc1[4] = {0.f, 0.f, 0.f, 0.f}, acc2[4] = {0.f, 0.f, 0.f, 0.f};

  for (int t = gw; t < N_TILES; t += nw) {
    const float* rp = feat + (size_t)(t * 16 + r) * CC + kb;
    if (t + nw < N_TILES)
      __builtin_prefetch(feat + (size_t)((t + nw) * 16 + r) * CC + kb, 0, 1);
    v16h A0, A1;
    if (t < T_CACHE) {           // head region: keep resident for pass 2
      A0 = make_afrag<false>(rp);
      A1 = make_afrag<false>(rp + 32);
    } else {                     // tail: stream non-temporally
      A0 = make_afrag<true>(rp);
      A1 = make_afrag<true>(rp + 32);
    }
#pragma unroll
    for (int ct = 0; ct < 4; ++ct) {
      v8f c = {};
      c = WMMA_F16(A0, B1[2 * ct + 0], c);
      c = WMMA_F16(A1, B1[2 * ct + 1], c);
#pragma unroll
      for (int r8 = 0; r8 < 8; ++r8) {
        const float v = c[r8];
        acc1[ct] += v;
        acc2[ct] += v * v;
      }
    }
  }
  // Lane l and l+16 hold the same column (different row halves): fold.
#pragma unroll
  for (int ct = 0; ct < 4; ++ct) {
    acc1[ct] += __shfl_down(acc1[ct], 16, 32);
    acc2[ct] += __shfl_down(acc2[ct], 16, 32);
  }
  if (lane < 16) {
#pragma unroll
    for (int ct = 0; ct < 4; ++ct) {
      atomicAdd(&sums[ct * 16 + lane], acc1[ct]);
      atomicAdd(&sums[64 + ct * 16 + lane], acc2[ct]);
    }
  }
}

// Fold stats into per-channel scale/shift: h_bn = g*s + t.
__global__ void bn_stats(const float* __restrict__ sums, const float* __restrict__ gamma,
                         const float* __restrict__ beta, float* __restrict__ st) {
  const int c = threadIdx.x;  // 64 threads
  const float invN = 1.0f / (float)N_PTS;
  const float mean = sums[c] * invN;
  const float var = sums[64 + c] * invN - mean * mean;
  const float s = gamma[c] * rsqrtf(var + 1e-3f);
  st[c] = s;
  st[64 + c] = beta[c] - mean * s;
}

// Pass 2: fused BN+ReLU heads + all three losses.
__global__ __launch_bounds__(128) void pass2_loss(
    const float* __restrict__ feat, const float* __restrict__ coord,
    const int* __restrict__ segment, const int* __restrict__ instance,
    const float* __restrict__ centroid, const float* __restrict__ b2,
    const float* __restrict__ bseg, const float* __restrict__ cw,
    const _Float16* __restrict__ frags, const float* __restrict__ st,
    float* __restrict__ acc) {
  const int lane = threadIdx.x & 31;
  const int wave = threadIdx.x >> 5;
  const int gw = (blockIdx.x * blockDim.x + threadIdx.x) >> 5;
  const int nw = (gridDim.x * blockDim.x) >> 5;
  const int r = lane & 15;
  const int kb = (lane >> 4) * 8;

  __shared__ _Float16 sh_h[4][16 * H_STRIDE];
  __shared__ float sh_l[4][16 * L_STRIDE];
  __shared__ float sh_b[4][16 * B_STRIDE];
  __shared__ float s_bseg[KCLS];
  __shared__ float s_b2[3];
  if (threadIdx.x < KCLS) s_bseg[threadIdx.x] = bseg[threadIdx.x];
  if (threadIdx.x < 3) s_b2[threadIdx.x] = b2[threadIdx.x];
  __syncthreads();

  v16h B1[8], BS[4], BW2[2];
#pragma unroll
  for (int f = 0; f < 8; ++f) B1[f] = *(const v16h*)(frags + f * 512 + lane * 16);
#pragma unroll
  for (int f = 0; f < 4; ++f) BS[f] = *(const v16h*)(frags + 4096 + f * 512 + lane * 16);
#pragma unroll
  for (int f = 0; f < 2; ++f) BW2[f] = *(const v16h*)(frags + 6144 + f * 512 + lane * 16);

  float sc[4], shf[4];
#pragma unroll
  for (int ct = 0; ct < 4; ++ct) {
    const int n = ct * 16 + r;
    sc[ct] = st[n];
    shf[ct] = st[64 + n];
  }

  float a_num = 0.f, a_den = 0.f, a_l1 = 0.f, a_cos = 0.f, a_m = 0.f;

  for (int t = gw; t < N_TILES; t += nw) {
    const float* rp = feat + (size_t)(t * 16 + r) * CC + kb;
    if (t + nw < N_TILES)
      __builtin_prefetch(feat + (size_t)((t + nw) * 16 + r) * CC + kb, 0, 1);
    v16h A0, A1;
    if (t < T_CACHE) {           // resident head: RT loads -> L2 hits
      A0 = make_afrag<false>(rp);
      A1 = make_afrag<false>(rp + 32);
    } else {                     // tail: stream non-temporally
      A0 = make_afrag<true>(rp);
      A1 = make_afrag<true>(rp + 32);
    }

    // g = feat @ W1 ; BN scale/shift + ReLU ; stash f16 tile row-major in LDS
#pragma unroll
    for (int ct = 0; ct < 4; ++ct) {
      v8f c = {};
      c = WMMA_F16(A0, B1[2 * ct + 0], c);
      c = WMMA_F16(A1, B1[2 * ct + 1], c);
      _Float16* hp = &sh_h[wave][(lane >> 4) * 8 * H_STRIDE + ct * 16 + r];
#pragma unroll
      for (int r8 = 0; r8 < 8; ++r8) {
        const float h = fmaxf(c[r8] * sc[ct] + shf[ct], 0.0f);
        hp[r8 * H_STRIDE] = (_Float16)h;
      }
    }

    // bias_pred = h @ W2 (W2 zero-padded to 64x16)
    const _Float16* hrp = &sh_h[wave][r * H_STRIDE + kb];
    v16h HA0 = make_hfrag(hrp);
    v16h HA1 = make_hfrag(hrp + 32);
    {
      v8f c = {};
      c = WMMA_F16(HA0, BW2[0], c);
      c = WMMA_F16(HA1, BW2[1], c);
      float* bp = &sh_b[wave][(lane >> 4) * 8 * B_STRIDE + r];
#pragma unroll
      for (int r8 = 0; r8 < 8; ++r8) bp[r8 * B_STRIDE] = c[r8];
    }

    // logit_pred = feat @ Wseg (zero-padded to 64x32)
#pragma unroll
    for (int ct = 0; ct < 2; ++ct) {
      v8f c = {};
      c = WMMA_F16(A0, BS[2 * ct + 0], c);
      c = WMMA_F16(A1, BS[2 * ct + 1], c);
      float* lp = &sh_l[wave][(lane >> 4) * 8 * L_STRIDE + ct * 16 + r];
#pragma unroll
      for (int r8 = 0; r8 < 8; ++r8) lp[r8 * L_STRIDE] = c[r8];
    }

    // Per-row losses: 16 lanes, one row each (dynamic indexing stays in LDS).
    if (lane < 16) {
      const int row = t * 16 + lane;
      const float* lp = &sh_l[wave][lane * L_STRIDE];
      float mx = -3.0e38f;
#pragma unroll
      for (int k = 0; k < KCLS; ++k) mx = fmaxf(mx, lp[k] + s_bseg[k]);
      float se = 0.f;
#pragma unroll
      for (int k = 0; k < KCLS; ++k) se += __expf(lp[k] + s_bseg[k] - mx);
      const int tgt = __builtin_nontemporal_load(segment + row);
      const float vt = (tgt == -1) ? 0.f : 1.f;
      const int tc = (tgt == -1) ? 0 : tgt;
      const float w = cw[tc] * vt;
      const float nll = mx + __logf(se) - (lp[tc] + s_bseg[tc]);
      a_num += w * nll;
      a_den += w;

      const float bx = sh_b[wave][lane * B_STRIDE + 0] + s_b2[0];
      const float by = sh_b[wave][lane * B_STRIDE + 1] + s_b2[1];
      const float bz = sh_b[wave][lane * B_STRIDE + 2] + s_b2[2];
      const float gx = __builtin_nontemporal_load(centroid + 3 * row + 0) -
                       __builtin_nontemporal_load(coord + 3 * row + 0);
      const float gy = __builtin_nontemporal_load(centroid + 3 * row + 1) -
                       __builtin_nontemporal_load(coord + 3 * row + 1);
      const float gz = __builtin_nontemporal_load(centroid + 3 * row + 2) -
                       __builtin_nontemporal_load(coord + 3 * row + 2);
      const float m = (__builtin_nontemporal_load(instance + row) == -1) ? 0.f : 1.f;
      a_l1 += m * (fabsf(bx - gx) + fabsf(by - gy) + fabsf(bz - gz));
      const float bn = sqrtf(bx * bx + by * by + bz * bz) + 1e-8f;
      const float gn = sqrtf(gx * gx + gy * gy + gz * gz) + 1e-8f;
      a_cos += m * (-(bx * gx + by * gy + bz * gz) / (bn * gn));
      a_m += m;
    }
  }

  a_num = wave_sum(a_num);
  a_den = wave_sum(a_den);
  a_l1 = wave_sum(a_l1);
  a_cos = wave_sum(a_cos);
  a_m = wave_sum(a_m);
  if (lane == 0) {
    atomicAdd(&acc[0], a_num);
    atomicAdd(&acc[1], a_den);
    atomicAdd(&acc[2], a_l1);
    atomicAdd(&acc[3], a_cos);
    atomicAdd(&acc[4], a_m);
  }
}

__global__ void finalize_loss(const float* __restrict__ acc, float* __restrict__ out) {
  const float seg = acc[0] / acc[1];
  const float l1 = acc[2] / (acc[4] + 1e-8f);
  const float cs = acc[3] / (acc[4] + 1e-8f);
  out[0] = seg + l1 + cs;
  out[1] = seg;
  out[2] = l1;
  out[3] = cs;
}

// ---------------------------------------------------------------------------
extern "C" void kernel_launch(void* const* d_in, const int* in_sizes, int n_in,
                              void* d_out, int out_size, void* d_ws, size_t ws_size,
                              hipStream_t stream) {
  const float* feat     = (const float*)d_in[0];
  const float* coord    = (const float*)d_in[1];
  const int*   segment  = (const int*)d_in[2];
  const int*   instance = (const int*)d_in[3];
  const float* centroid = (const float*)d_in[4];
  const float* W1       = (const float*)d_in[5];
  // d_in[6] = b1: cancels in BatchNorm (mean-subtracted), intentionally unused.
  const float* gamma    = (const float*)d_in[7];
  const float* beta     = (const float*)d_in[8];
  const float* W2       = (const float*)d_in[9];
  const float* b2       = (const float*)d_in[10];
  const float* Wseg     = (const float*)d_in[11];
  const float* bseg     = (const float*)d_in[12];
  const float* cw       = (const float*)d_in[13];
  float* out = (float*)d_out;

  float* wsf = (float*)d_ws;
  float* sums = wsf;          // [0..127]  colsum / colsumsq of g
  float* st   = wsf + 128;    // [128..255] scale / shift
  float* acc  = wsf + 256;    // [256..263] loss accumulators
  _Float16* frags = (_Float16*)((char*)d_ws + 4096);  // 7168 f16 B-fragments

  hipLaunchKernelGGL(init_ws, dim3(1), dim3(128), 0, stream, wsf);
  hipLaunchKernelGGL(prep_frags, dim3(1), dim3(32), 0, stream, W1, Wseg, W2, frags);
  hipLaunchKernelGGL(pass1_stats, dim3(1024), dim3(128), 0, stream, feat, frags, sums);
  hipLaunchKernelGGL(bn_stats, dim3(1), dim3(64), 0, stream, sums, gamma, beta, st);
  hipLaunchKernelGGL(pass2_loss, dim3(2048), dim3(128), 0, stream, feat, coord, segment,
                     instance, centroid, b2, bseg, cw, frags, st, acc);
  hipLaunchKernelGGL(finalize_loss, dim3(1), dim3(1), 0, stream, acc, out);
}
